// encoder_layer_26560077758730
// MI455X (gfx1250) — matrix operand order repaired and checked by reading, not compiled, not run on hardware
//
#include <hip/hip_runtime.h>

// ---------------- CDNA5 (gfx1250) wave32 WMMA types ----------------
typedef __attribute__((ext_vector_type(16))) __bf16 v16bf;
typedef __attribute__((ext_vector_type(8)))  __bf16 v8bf;
typedef __attribute__((ext_vector_type(8)))  float  v8f;
typedef __attribute__((ext_vector_type(4)))  unsigned int v4u;
typedef __attribute__((ext_vector_type(8)))  int v8i;
typedef __attribute__((ext_vector_type(4)))  int v4i;

// Tensor Data Mover availability (device pass only; host pass falls back, harmless)
#if defined(__has_builtin)
#if __has_builtin(__builtin_amdgcn_tensor_load_to_lds)
#define USE_TDM 1
#endif
#endif
#ifndef USE_TDM
#define USE_TDM 0
#endif

// A-fragment (16x32 bf16, row-major source, row = lane%16):
//   a[i]: i in [0,16) -> K = half*16 + i
// => two contiguous 8-element (16B) runs per lane.
__device__ __forceinline__ v16bf make_a_frag(v8bf lo, v8bf hi) {
  return __builtin_shufflevector(lo, hi, 0, 1, 2, 3, 4, 5, 6, 7,
                                 8, 9, 10, 11, 12, 13, 14, 15);
}
__device__ __forceinline__ v16bf load_a_frag(const __bf16* rowp, int hf) {
  v8bf lo = *(const v8bf*)(rowp + hf * 16);
  v8bf hi = *(const v8bf*)(rowp + hf * 16 + 8);
  return make_a_frag(lo, hi);
}

#if USE_TDM
// Issue one 2D TDM tile load: (tile_d1=128 rows) x (tile_d0=32 bf16) from a
// row-major tensor with row stride `stride_elems`, into LDS at `lds_addr`.
// D# packing per CDNA5 ISA (data_size=1 -> 2B).
__device__ __forceinline__ void tdm_load_tile2d(unsigned lds_addr,
                                                const __bf16* gptr,
                                                unsigned tensor_d0,
                                                unsigned tensor_d1,
                                                unsigned long long stride_elems) {
  unsigned long long ga = (unsigned long long)(size_t)gptr;
  v4u g0;
  g0[0] = 1u;                                            // count=1 (valid user D#)
  g0[1] = lds_addr;                                      // LDS byte address
  g0[2] = (unsigned)(ga & 0xffffffffu);                  // global_addr[31:0]
  g0[3] = (unsigned)((ga >> 32) & 0x01ffffffu) | (2u << 30);  // addr[56:32] | type=2
  v8i g1;
  g1[0] = (int)(1u << 16);                               // data_size=1 (2 bytes)
  g1[1] = (int)((tensor_d0 & 0xffffu) << 16);            // tensor_dim0[15:0]
  g1[2] = (int)(((tensor_d0 >> 16) & 0xffffu) |
                ((tensor_d1 & 0xffffu) << 16));          // td0[31:16] | td1[15:0]
  g1[3] = (int)(((tensor_d1 >> 16) & 0xffffu) |
                (32u << 16));                            // td1[31:16] | tile_dim0=32
  g1[4] = (int)(128u);                                   // tile_dim1=128, tile_dim2=0
  g1[5] = (int)(stride_elems & 0xffffffffull);           // tensor_dim0_stride[31:0]
  g1[6] = (int)((stride_elems >> 32) & 0xffffull);       // stride[47:32] | td1_stride=0
  g1[7] = 0;
  v4i z4 = {0, 0, 0, 0};
#if __clang_major__ >= 23
  v8i z8 = {0, 0, 0, 0, 0, 0, 0, 0};
  __builtin_amdgcn_tensor_load_to_lds(g0, g1, z4, z4, z8, 0);
#else
  __builtin_amdgcn_tensor_load_to_lds(g0, g1, z4, z4, 0);
#endif
}
#endif  // USE_TDM

// ---------------- elementwise f32 -> bf16 convert ----------------
__global__ void cvt_bf16_kernel(const float* __restrict__ in,
                                __bf16* __restrict__ out, int n) {
  int i = blockIdx.x * blockDim.x + threadIdx.x;
  int stride = gridDim.x * blockDim.x;
  for (; i < n; i += stride) out[i] = (__bf16)in[i];
}

// ---------------- tiled transpose + convert: in[K x N] -> out[N x K] ----------------
__global__ __launch_bounds__(256)
void transpose_cvt_kernel(const float* __restrict__ in, __bf16* __restrict__ out,
                          int K, int N) {
  __shared__ float tile[32][33];
  int x = blockIdx.x * 32 + threadIdx.x;  // col n
  int y = blockIdx.y * 32 + threadIdx.y;  // row k
#pragma unroll
  for (int i = 0; i < 32; i += 8)
    if (x < N && (y + i) < K)
      tile[threadIdx.y + i][threadIdx.x] = in[(size_t)(y + i) * N + x];
  __syncthreads();
  int nx = blockIdx.y * 32 + threadIdx.x;  // k
  int ny = blockIdx.x * 32 + threadIdx.y;  // n
#pragma unroll
  for (int i = 0; i < 32; i += 8)
    if (nx < K && (ny + i) < N)
      out[(size_t)(ny + i) * K + nx] = (__bf16)tile[threadIdx.x][threadIdx.y + i];
}

// ---------------- WMMA GEMM: C[M,N] = A[M,K](bf16,row) * Bt[N,K](bf16,row) + bias ----
// Block tile 128x128 (8 waves, each 32x64 = 2x4 WMMA fragments).
// TDM path: wave 0 DMAs the 128x32 A and B tiles into double-buffered LDS
// (TENSORcnt pipelined), all waves consume via ds_load_b128.
// MODE 0: bf16 out in (B,H,S,D) head layout            (Q, K projections)
// MODE 1: bf16 out in (B,H,D,S) transposed head layout (V projection)
// MODE 2: f32 out = gemm + bias + residual             (Wo proj, FFN2)
// MODE 3: bf16 out = relu(gemm + bias)                 (FFN1)
template <int MODE>
__global__ __launch_bounds__(256)
void gemm_bf16_wmma(const __bf16* __restrict__ A, const __bf16* __restrict__ Bt,
                    const float* __restrict__ bias, const float* __restrict__ res,
                    float* __restrict__ outF, __bf16* __restrict__ outB,
                    int M, int N, int K, int S) {
  const int lane = threadIdx.x & 31;
  const int wave = threadIdx.x >> 5;
  const int hf = lane >> 4;    // half-wave select
  const int l16 = lane & 15;
  const int wr = wave & 3;
  const int wc = wave >> 2;
  const int rowBase = blockIdx.y * 128 + wr * 32;
  const int colBase = blockIdx.x * 128 + wc * 64;

  v8f acc[2][4] = {};

#if USE_TDM
  __shared__ __attribute__((aligned(128))) __bf16 As[2][128 * 32];
  __shared__ __attribute__((aligned(128))) __bf16 Bs[2][128 * 32];
  const unsigned asb = (unsigned)(size_t)&As[0][0];
  const unsigned bsb = (unsigned)(size_t)&Bs[0][0];
  const __bf16* Ablk = A + (size_t)blockIdx.y * 128 * K;
  const __bf16* Bblk = Bt + (size_t)blockIdx.x * 128 * K;
  if (wave == 0) {  // prologue: stage first K-tile pair
    tdm_load_tile2d(asb, Ablk, (unsigned)K, (unsigned)M, (unsigned long long)K);
    tdm_load_tile2d(bsb, Bblk, (unsigned)K, (unsigned)N, (unsigned long long)K);
  }
  for (int kk = 0; kk < K; kk += 32) {
    const int cur = (kk >> 5) & 1;
    if (wave == 0) {
      if (kk + 32 < K) {  // stage next pair into the other buffer, then
        tdm_load_tile2d(asb + (unsigned)((cur ^ 1) * 8192), Ablk + kk + 32,
                        (unsigned)K, (unsigned)M, (unsigned long long)K);
        tdm_load_tile2d(bsb + (unsigned)((cur ^ 1) * 8192), Bblk + kk + 32,
                        (unsigned)K, (unsigned)N, (unsigned long long)K);
        __builtin_amdgcn_s_wait_tensorcnt(2);  // current pair complete
      } else {
        __builtin_amdgcn_s_wait_tensorcnt(0);
      }
    }
    __syncthreads();
    const __bf16* Ac = &As[cur][0];
    const __bf16* Bc = &Bs[cur][0];
    v16bf af[2], bfr[4];
#pragma unroll
    for (int rm = 0; rm < 2; ++rm) {
      const __bf16* rp = Ac + (size_t)(wr * 32 + rm * 16 + l16) * 32;
      v8bf lo = *(const v8bf*)(rp + hf * 16);
      v8bf hi = *(const v8bf*)(rp + hf * 16 + 8);
      af[rm] = make_a_frag(lo, hi);
    }
#pragma unroll
    for (int cn = 0; cn < 4; ++cn)
      bfr[cn] = *(const v16bf*)(Bc + (size_t)(wc * 64 + cn * 16 + l16) * 32 + hf * 16);
#pragma unroll
    for (int rm = 0; rm < 2; ++rm)
#pragma unroll
      for (int cn = 0; cn < 4; ++cn)
        acc[rm][cn] = __builtin_amdgcn_wmma_f32_16x16x32_bf16(
            false, af[rm], false, bfr[cn], (short)0, acc[rm][cn], false, false);
    __syncthreads();
  }
#else  // fallback: direct global fragment loads
  const __bf16* ap[2];
  const __bf16* bp[4];
#pragma unroll
  for (int rm = 0; rm < 2; ++rm)
    ap[rm] = A + (size_t)(rowBase + rm * 16 + l16) * K;
#pragma unroll
  for (int cn = 0; cn < 4; ++cn)
    bp[cn] = Bt + (size_t)(colBase + cn * 16 + l16) * K + hf * 16;
  for (int kk = 0; kk < K; kk += 32) {
    if (kk + 64 <= K) {
      __builtin_prefetch(ap[0] + kk + 64, 0, 1);
      __builtin_prefetch(bp[0] + kk + 64, 0, 1);
    }
    v16bf af[2], bfr[4];
#pragma unroll
    for (int rm = 0; rm < 2; ++rm) af[rm] = load_a_frag(ap[rm] + kk, hf);
#pragma unroll
    for (int cn = 0; cn < 4; ++cn) bfr[cn] = *(const v16bf*)(bp[cn] + kk);
#pragma unroll
    for (int rm = 0; rm < 2; ++rm)
#pragma unroll
      for (int cn = 0; cn < 4; ++cn)
        acc[rm][cn] = __builtin_amdgcn_wmma_f32_16x16x32_bf16(
            false, af[rm], false, bfr[cn], (short)0, acc[rm][cn], false, false);
  }
#endif

  // C/D layout: VGPR r -> row = half*8 + r, col = lane%16
#pragma unroll
  for (int rm = 0; rm < 2; ++rm) {
#pragma unroll
    for (int cn = 0; cn < 4; ++cn) {
      const int n = colBase + cn * 16 + l16;
      const float bn = bias[n];
#pragma unroll
      for (int r = 0; r < 8; ++r) {
        const int m = rowBase + rm * 16 + hf * 8 + r;
        float v = acc[rm][cn][r] + bn;
        if (MODE == 2) {
          outF[(size_t)m * N + n] = v + res[(size_t)m * N + n];
        } else if (MODE == 3) {
          outB[(size_t)m * N + n] = (__bf16)fmaxf(v, 0.0f);
        } else {
          const int b = m / S, s = m - b * S;
          const int hh = n >> 6, d = n & 63;
          if (MODE == 0)
            outB[(((size_t)b * 16 + hh) * S + s) * 64 + d] = (__bf16)v;
          else
            outB[(((size_t)b * 16 + hh) * 64 + d) * (size_t)S + s] = (__bf16)v;
        }
      }
    }
  }
}

// ---------------- flash attention (one wave = 16 query rows) ----------------
// Q,Kc: (B,H,S,64) bf16 ; Vt: (B,H,64,S) bf16 ; out: (B,S,1024) bf16
__global__ __launch_bounds__(128)
void flash_attn_kernel(const __bf16* __restrict__ Q, const __bf16* __restrict__ Kc,
                       const __bf16* __restrict__ Vt, const float* __restrict__ mask,
                       __bf16* __restrict__ out, int S) {
  const int lane = threadIdx.x & 31;
  const int wave = threadIdx.x >> 5;
  const int hf = lane >> 4;
  const int l16 = lane & 15;
  const int nqb = S / 64;
  const int qb = blockIdx.x % nqb;
  const int bh = blockIdx.x / nqb;
  const int b = bh >> 4;  // H = 16
  const int h = bh & 15;
  const int q0 = qb * 64 + wave * 16;

  const __bf16* Qb = Q + (size_t)bh * S * 64;
  const __bf16* Kb = Kc + (size_t)bh * S * 64;
  const __bf16* Vb = Vt + (size_t)bh * 64 * S;
  const float* mrow = mask + (size_t)b * S;

  __shared__ __attribute__((aligned(32))) __bf16 Pbuf[4][16 * 32];
  __bf16* P = Pbuf[wave];

  // Q tile fragments (16 rows x 64 depth = two K=32 fragments)
  v16bf qf[2];
  {
    const __bf16* qp = Qb + (size_t)(q0 + l16) * 64;
#pragma unroll
    for (int kd = 0; kd < 2; ++kd) qf[kd] = load_a_frag(qp + kd * 32, hf);
  }

  v8f acc[4] = {};
  float mrun[8], lrun[8], corr[8];
#pragma unroll
  for (int r = 0; r < 8; ++r) { mrun[r] = -1e30f; lrun[r] = 0.0f; }

  for (int j = 0; j < S; j += 32) {
    // scores: two 16x16 fragments over 32 keys, depth reduced over 64
    v8f sc[2];
#pragma unroll
    for (int c = 0; c < 2; ++c) {
      const __bf16* kp = Kb + (size_t)(j + c * 16 + l16) * 64 + hf * 16;
      v16bf k0 = *(const v16bf*)kp;         // depth 0..31
      v16bf k1 = *(const v16bf*)(kp + 32);  // depth 32..63
      v8f z = {};
      z = __builtin_amdgcn_wmma_f32_16x16x32_bf16(false, qf[0], false, k0,
                                                  (short)0, z, false, false);
      z = __builtin_amdgcn_wmma_f32_16x16x32_bf16(false, qf[1], false, k1,
                                                  (short)0, z, false, false);
      sc[c] = z;
    }

    float sv[2][8];
#pragma unroll
    for (int c = 0; c < 2; ++c) {
      const float mv = mrow[j + c * 16 + l16] * -1e12f;
#pragma unroll
      for (int r = 0; r < 8; ++r) sv[c][r] = sc[c][r] * 0.125f + mv;
    }

    // online softmax per row (rows live across 16 lanes of each half-wave)
#pragma unroll
    for (int r = 0; r < 8; ++r) {
      float rm = fmaxf(sv[0][r], sv[1][r]);
#pragma unroll
      for (int mlw = 1; mlw < 16; mlw <<= 1) rm = fmaxf(rm, __shfl_xor(rm, mlw, 32));
      const float mnew = fmaxf(mrun[r], rm);
      corr[r] = __expf(mrun[r] - mnew);
      const float p0 = __expf(sv[0][r] - mnew);
      const float p1 = __expf(sv[1][r] - mnew);
      float rs = p0 + p1;
#pragma unroll
      for (int mlw = 1; mlw < 16; mlw <<= 1) rs += __shfl_xor(rs, mlw, 32);
      lrun[r] = lrun[r] * corr[r] + rs;
      mrun[r] = mnew;
      sv[0][r] = p0;
      sv[1][r] = p1;
    }
#pragma unroll
    for (int f = 0; f < 4; ++f)
#pragma unroll
      for (int r = 0; r < 8; ++r) acc[f][r] *= corr[r];

    // stage P (16x32) through LDS: C-layout -> A-layout
#pragma unroll
    for (int c = 0; c < 2; ++c)
#pragma unroll
      for (int r = 0; r < 8; ++r)
        P[(hf * 8 + r) * 32 + c * 16 + l16] = (__bf16)sv[c][r];
    __syncthreads();
    v16bf pa;
    {
      v8bf lo = *(const v8bf*)(P + l16 * 32 + hf * 16);
      v8bf hi = *(const v8bf*)(P + l16 * 32 + hf * 16 + 8);
      pa = make_a_frag(lo, hi);
    }
#pragma unroll
    for (int f = 0; f < 4; ++f) {
      const __bf16* vp = Vb + (size_t)(f * 16 + l16) * S + j + hf * 16;
      v16bf vv = *(const v16bf*)vp;
      acc[f] = __builtin_amdgcn_wmma_f32_16x16x32_bf16(false, pa, false, vv,
                                                       (short)0, acc[f], false, false);
    }
    __syncthreads();
  }

#pragma unroll
  for (int f = 0; f < 4; ++f) {
#pragma unroll
    for (int r = 0; r < 8; ++r) {
      const float v = acc[f][r] / lrun[r];
      const int srow = q0 + hf * 8 + r;
      const int col = h * 64 + f * 16 + l16;
      out[((size_t)b * S + srow) * 1024 + col] = (__bf16)v;
    }
  }
}

// ---------------- LayerNorm: one block per row of E floats ----------------
__global__ __launch_bounds__(256)
void layernorm_kernel(const float* __restrict__ in, const float* __restrict__ g,
                      const float* __restrict__ be, float* __restrict__ outF,
                      __bf16* __restrict__ outB, int E) {
  const int row = blockIdx.x;
  const float* x = in + (size_t)row * E;
  float s = 0.f, s2 = 0.f;
  for (int i = threadIdx.x; i < E; i += 256) {
    const float v = x[i];
    s += v;
    s2 += v * v;
  }
#pragma unroll
  for (int m = 1; m < 32; m <<= 1) {
    s += __shfl_xor(s, m, 32);
    s2 += __shfl_xor(s2, m, 32);
  }
  __shared__ float sh[2][8];
  if ((threadIdx.x & 31) == 0) {
    sh[0][threadIdx.x >> 5] = s;
    sh[1][threadIdx.x >> 5] = s2;
  }
  __syncthreads();
  s = sh[0][threadIdx.x & 7];
  s2 = sh[1][threadIdx.x & 7];
#pragma unroll
  for (int m = 1; m < 8; m <<= 1) {
    s += __shfl_xor(s, m, 32);
    s2 += __shfl_xor(s2, m, 32);
  }
  const float mu = s * (1.0f / E);
  const float var = s2 * (1.0f / E) - mu * mu;
  const float inv = rsqrtf(var + 1e-9f);
  for (int i = threadIdx.x; i < E; i += 256) {
    const float y = (x[i] - mu) * inv * g[i] + be[i];
    outF[(size_t)row * E + i] = y;
    if (outB) outB[(size_t)row * E + i] = (__bf16)y;
  }
}

// ---------------- host launch ----------------
extern "C" void kernel_launch(void* const* d_in, const int* in_sizes, int n_in,
                              void* d_out, int out_size, void* d_ws, size_t ws_size,
                              hipStream_t stream) {
  (void)in_sizes; (void)n_in; (void)out_size; (void)ws_size;
  const float* x    = (const float*)d_in[0];
  const float* mask = (const float*)d_in[1];
  const float* wq_w = (const float*)d_in[2];
  const float* wq_b = (const float*)d_in[3];
  const float* wk_w = (const float*)d_in[4];
  const float* wk_b = (const float*)d_in[5];
  const float* wv_w = (const float*)d_in[6];
  const float* wv_b = (const float*)d_in[7];
  const float* wo_w = (const float*)d_in[8];
  const float* wo_b = (const float*)d_in[9];
  const float* f1_w = (const float*)d_in[10];
  const float* f1_b = (const float*)d_in[11];
  const float* f2_w = (const float*)d_in[12];
  const float* f2_b = (const float*)d_in[13];
  const float* ln1_g = (const float*)d_in[14];
  const float* ln1_b = (const float*)d_in[15];
  const float* ln2_g = (const float*)d_in[16];
  const float* ln2_b = (const float*)d_in[17];

  const int B = 2, S = 2048, E = 1024, F = 4096, M = B * S;
  char* ws = (char*)d_ws;
  const size_t MB = 1024 * 1024;

  __bf16* wq_t = (__bf16*)(ws + 0 * MB);
  __bf16* wk_t = (__bf16*)(ws + 2 * MB);
  __bf16* wv_t = (__bf16*)(ws + 4 * MB);
  __bf16* wo_t = (__bf16*)(ws + 6 * MB);
  __bf16* f1_t = (__bf16*)(ws + 8 * MB);    // 8 MB
  __bf16* f2_t = (__bf16*)(ws + 16 * MB);   // 8 MB
  __bf16* x_bf = (__bf16*)(ws + 24 * MB);   // 8 MB
  __bf16* attn_bf = x_bf;                   // reuse: x_bf dead after QKV GEMMs
  __bf16* ffn_bf = x_bf;                    // reuse: 32 MB span over x_bf,q,k,vt
  __bf16* q_bf  = (__bf16*)(ws + 32 * MB);  // 8 MB
  __bf16* k_bf  = (__bf16*)(ws + 40 * MB);  // 8 MB
  __bf16* vt_bf = (__bf16*)(ws + 48 * MB);  // 8 MB
  float*  tmp   = (float*)(ws + 56 * MB);   // 16 MB (LN1 input, later LN2 input)
  float*  h_f   = (float*)(ws + 72 * MB);   // 16 MB
  __bf16* h_bf  = (__bf16*)(ws + 88 * MB);  // 8 MB  -> 96 MB total

  // 1) precision conversion (bandwidth-trivial; weights then live in L2)
  cvt_bf16_kernel<<<2048, 256, 0, stream>>>(x, x_bf, M * E);
  dim3 tb(32, 8);
  transpose_cvt_kernel<<<dim3(E / 32, E / 32), tb, 0, stream>>>(wq_w, wq_t, E, E);
  transpose_cvt_kernel<<<dim3(E / 32, E / 32), tb, 0, stream>>>(wk_w, wk_t, E, E);
  transpose_cvt_kernel<<<dim3(E / 32, E / 32), tb, 0, stream>>>(wv_w, wv_t, E, E);
  transpose_cvt_kernel<<<dim3(E / 32, E / 32), tb, 0, stream>>>(wo_w, wo_t, E, E);
  transpose_cvt_kernel<<<dim3(F / 32, E / 32), tb, 0, stream>>>(f1_w, f1_t, E, F);
  transpose_cvt_kernel<<<dim3(E / 32, F / 32), tb, 0, stream>>>(f2_w, f2_t, F, E);

  // 2) QKV projections (WMMA bf16, TDM-staged tiles)
  dim3 gE(E / 128, M / 128);
  gemm_bf16_wmma<0><<<gE, 256, 0, stream>>>(x_bf, wq_t, wq_b, nullptr, nullptr, q_bf, M, E, E, S);
  gemm_bf16_wmma<0><<<gE, 256, 0, stream>>>(x_bf, wk_t, wk_b, nullptr, nullptr, k_bf, M, E, E, S);
  gemm_bf16_wmma<1><<<gE, 256, 0, stream>>>(x_bf, wv_t, wv_b, nullptr, nullptr, vt_bf, M, E, E, S);

  // 3) flash attention (scale 1/sqrt(64) + mask fused)
  flash_attn_kernel<<<B * 16 * (S / 64), 128, 0, stream>>>(q_bf, k_bf, vt_bf, mask, attn_bf, S);

  // 4) output projection + residual -> LN1
  gemm_bf16_wmma<2><<<gE, 256, 0, stream>>>(attn_bf, wo_t, wo_b, x, tmp, nullptr, M, E, E, S);
  layernorm_kernel<<<M, 256, 0, stream>>>(tmp, ln1_g, ln1_b, h_f, h_bf, E);

  // 5) FFN: relu(h@f1+b1)@f2+b2 + residual -> LN2 -> d_out
  gemm_bf16_wmma<3><<<dim3(F / 128, M / 128), 256, 0, stream>>>(h_bf, f1_t, f1_b, nullptr, nullptr, ffn_bf, M, F, E, S);
  gemm_bf16_wmma<2><<<gE, 256, 0, stream>>>(ffn_bf, f2_t, f2_b, h_f, tmp, nullptr, M, E, F, S);
  layernorm_kernel<<<M, 256, 0, stream>>>(tmp, ln2_g, ln2_b, (float*)d_out, nullptr, E);
}
